// LinearAttention_23467701305647
// MI455X (gfx1250) — matrix operand order repaired and checked by reading, not compiled, not run on hardware
//
#include <hip/hip_runtime.h>
#include <hip/hip_bf16.h>
#include <math.h>
#include <stdint.h>

#define DIMC   512
#define HEADS  8
#define SEQ    4096
#define BATCH  8
#define QKV_ROWS 1536

#define BK 32
#define APITCH 40   // bf16 elems per LDS row: 32 + 8 pad (80B pitch, 16B-aligned)

typedef __attribute__((ext_vector_type(16))) __bf16 v16bf;
typedef __attribute__((ext_vector_type(8)))  __bf16 v8bf;
typedef __attribute__((ext_vector_type(8)))  float  v8f;
typedef __attribute__((ext_vector_type(4)))  uint32_t v4u;
typedef __attribute__((ext_vector_type(8)))  uint32_t v8u;

union FragBf { v16bf v; v8bf h[2]; };
union PackBf { __bf16 b[2]; uint32_t u; };

// low 32 bits of a generic pointer to __shared__ == LDS byte offset (ISA 10.2)
__device__ inline uint32_t lds_off(const void* p) { return (uint32_t)(uintptr_t)p; }

// ---------------------------------------------------------------------------
// GEMM1: qkv[b] = Wqkv(1536x512) @ x[b](512x4096); fp32 in -> bf16 out
// BM=128, BN=64, BK=32; double-buffered LDS, one barrier per K-step
// ---------------------------------------------------------------------------
__global__ __launch_bounds__(256)
void k_gemm_qkv(const float* __restrict__ W, const float* __restrict__ X,
                __bf16* __restrict__ QKV)
{
  __shared__ __bf16 As[2][128][APITCH];
  __shared__ __bf16 Bs[2][64][APITCH];
  const int tid = threadIdx.x, lane = tid & 31, wave = tid >> 5;
  const int n0 = blockIdx.x * 64;
  const int m0 = blockIdx.y * 128;
  const int b  = blockIdx.z;
  const float* Xb = X + (size_t)b * DIMC * SEQ;
  __bf16* Ob = QKV + (size_t)b * QKV_ROWS * SEQ;

  const int wm2 = (wave & 3) * 2, wn2 = (wave >> 2) * 2;
  v8f c00 = {}, c01 = {}, c10 = {}, c11 = {};

  const int ar = tid >> 2, ac = (tid & 3) * 8;
  const int kp = tid >> 4, ng = (tid & 15) * 4;

  auto load_tiles = [&](int buf, int kt) {
    #pragma unroll
    for (int rr = 0; rr < 2; ++rr) {   // A: fp32 -> bf16, K-contiguous
      const float* src = W + (size_t)(m0 + ar + rr*64) * DIMC + kt + ac;
      float4 f0 = ((const float4*)src)[0];
      float4 f1 = ((const float4*)src)[1];
      v8bf t;
      t[0]=(__bf16)f0.x; t[1]=(__bf16)f0.y; t[2]=(__bf16)f0.z; t[3]=(__bf16)f0.w;
      t[4]=(__bf16)f1.x; t[5]=(__bf16)f1.y; t[6]=(__bf16)f1.z; t[7]=(__bf16)f1.w;
      *(v8bf*)&As[buf][ar + rr*64][ac] = t;
    }
    { // B: pack (k,k+1) bf16 pairs -> b32 LDS stores (transpose)
      const float* s0 = Xb + (size_t)(kt + 2*kp)     * SEQ + n0 + ng;
      const float* s1 = Xb + (size_t)(kt + 2*kp + 1) * SEQ + n0 + ng;
      float4 f0 = *(const float4*)s0;
      float4 f1 = *(const float4*)s1;
      float a0[4] = {f0.x, f0.y, f0.z, f0.w};
      float a1[4] = {f1.x, f1.y, f1.z, f1.w};
      #pragma unroll
      for (int j = 0; j < 4; ++j) {
        PackBf p; p.b[0] = (__bf16)a0[j]; p.b[1] = (__bf16)a1[j];
        *(uint32_t*)&Bs[buf][ng + j][2*kp] = p.u;
      }
    }
  };

  auto compute = [&](int buf) {
    const int l15 = lane & 15, hi = lane >> 4;
    FragBf a0, a1, fb0, fb1;
    a0.h[0] = *(const v8bf*)&As[buf][(wm2+0)*16 + l15][hi*8];
    a0.h[1] = *(const v8bf*)&As[buf][(wm2+0)*16 + l15][hi*8 + 16];
    a1.h[0] = *(const v8bf*)&As[buf][(wm2+1)*16 + l15][hi*8];
    a1.h[1] = *(const v8bf*)&As[buf][(wm2+1)*16 + l15][hi*8 + 16];
    fb0.h[0] = *(const v8bf*)&Bs[buf][(wn2+0)*16 + l15][hi*8];
    fb0.h[1] = *(const v8bf*)&Bs[buf][(wn2+0)*16 + l15][hi*8 + 16];
    fb1.h[0] = *(const v8bf*)&Bs[buf][(wn2+1)*16 + l15][hi*8];
    fb1.h[1] = *(const v8bf*)&Bs[buf][(wn2+1)*16 + l15][hi*8 + 16];
    c00 = __builtin_amdgcn_wmma_f32_16x16x32_bf16(false, a0.v, false, fb0.v, (short)0, c00, false, false);
    c01 = __builtin_amdgcn_wmma_f32_16x16x32_bf16(false, a0.v, false, fb1.v, (short)0, c01, false, false);
    c10 = __builtin_amdgcn_wmma_f32_16x16x32_bf16(false, a1.v, false, fb0.v, (short)0, c10, false, false);
    c11 = __builtin_amdgcn_wmma_f32_16x16x32_bf16(false, a1.v, false, fb1.v, (short)0, c11, false, false);
  };

  load_tiles(0, 0);
  int p = 0;
  for (int i = 0; i < DIMC / BK; ++i) {
    __syncthreads();
    if (i + 1 < DIMC / BK) load_tiles(1 - p, (i + 1) * BK);
    compute(p);
    p ^= 1;
  }

  const int l15 = lane & 15, hi8 = (lane >> 4) * 8;
  #pragma unroll
  for (int i = 0; i < 8; ++i) {
    int r0 = m0 + (wm2+0)*16 + hi8 + i;
    int r1 = m0 + (wm2+1)*16 + hi8 + i;
    Ob[(size_t)r0 * SEQ + n0 + (wn2+0)*16 + l15] = (__bf16)c00[i];
    Ob[(size_t)r0 * SEQ + n0 + (wn2+1)*16 + l15] = (__bf16)c01[i];
    Ob[(size_t)r1 * SEQ + n0 + (wn2+0)*16 + l15] = (__bf16)c10[i];
    Ob[(size_t)r1 * SEQ + n0 + (wn2+1)*16 + l15] = (__bf16)c11[i];
  }
}

// ---------------------------------------------------------------------------
// Softmax over head-dim d (64) for q, times SCALE; in-place on qkv rows 0..511
// ---------------------------------------------------------------------------
__global__ __launch_bounds__(256)
void k_softmax_q(__bf16* __restrict__ QKV)
{
  int t = blockIdx.x * blockDim.x + threadIdx.x;
  int n = t & (SEQ - 1);
  int h = (t >> 12) & 7;
  int b = t >> 15;
  __bf16* base = QKV + (size_t)b * QKV_ROWS * SEQ + (size_t)(h * 64) * SEQ + n;
  float v[64];
  float m = -1e30f;
  #pragma unroll
  for (int d = 0; d < 64; ++d) { v[d] = (float)base[(size_t)d * SEQ]; m = fmaxf(m, v[d]); }
  float s = 0.f;
  #pragma unroll
  for (int d = 0; d < 64; ++d) { v[d] = __expf(v[d] - m); s += v[d]; }
  float inv = 0.125f / s;                          // SCALE = 64^-0.5
  #pragma unroll
  for (int d = 0; d < 64; ++d) base[(size_t)d * SEQ] = (__bf16)(v[d] * inv);
}

// ---------------------------------------------------------------------------
// Softmax over sequence n (4096) for k; in-place on qkv rows 512..1023
// ---------------------------------------------------------------------------
__global__ __launch_bounds__(256)
void k_softmax_k(__bf16* __restrict__ QKV)
{
  __shared__ float red[256];
  int row = blockIdx.x;                // b*512 + c
  int b = row >> 9, c = row & 511;
  __bf16* base = QKV + (size_t)b * QKV_ROWS * SEQ + (size_t)(512 + c) * SEQ;
  int tid = threadIdx.x;
  float m = -1e30f;
  for (int i = tid; i < SEQ; i += 256) m = fmaxf(m, (float)base[i]);
  red[tid] = m; __syncthreads();
  for (int s = 128; s > 0; s >>= 1) { if (tid < s) red[tid] = fmaxf(red[tid], red[tid + s]); __syncthreads(); }
  m = red[0]; __syncthreads();
  float sum = 0.f;
  for (int i = tid; i < SEQ; i += 256) sum += __expf((float)base[i] - m);
  red[tid] = sum; __syncthreads();
  for (int s = 128; s > 0; s >>= 1) { if (tid < s) red[tid] += red[tid + s]; __syncthreads(); }
  float inv = 1.0f / red[0];
  for (int i = tid; i < SEQ; i += 256) base[i] = (__bf16)(__expf((float)base[i] - m) * inv);
}

// ---------------------------------------------------------------------------
// ctx partials: split-K x8, async global->LDS copies (ASYNCcnt) + double buffer
// ---------------------------------------------------------------------------
__global__ __launch_bounds__(256)
void k_context(const __bf16* __restrict__ QKV, float* __restrict__ PART)
{
  __shared__ __bf16 Ks[2][64][APITCH];
  __shared__ __bf16 Vs[2][64][APITCH];
  const int tid = threadIdx.x, lane = tid & 31, wave = tid >> 5;
  const int bh = blockIdx.x >> 3;      // b*8 + h
  const int split = blockIdx.x & 7;
  const int b = bh >> 3, h = bh & 7;
  const __bf16* Kp = QKV + (size_t)b * QKV_ROWS * SEQ + (size_t)(512  + h*64) * SEQ;
  const __bf16* Vp = QKV + (size_t)b * QKV_ROWS * SEQ + (size_t)(1024 + h*64) * SEQ;
  const int wm = wave & 3, wn = (wave >> 2) * 2;
  v8f c0 = {}, c1 = {};
  const int r = tid >> 2, cg = (tid & 3) * 8;
  const uint32_t ldsK[2] = { lds_off(&Ks[0][r][cg]), lds_off(&Ks[1][r][cg]) };
  const uint32_t ldsV[2] = { lds_off(&Vs[0][r][cg]), lds_off(&Vs[1][r][cg]) };

  auto issue_async = [&](int buf, int kt) {
    unsigned long long gK = (unsigned long long)(uintptr_t)(Kp + (size_t)r * SEQ + kt + cg);
    unsigned long long gV = (unsigned long long)(uintptr_t)(Vp + (size_t)r * SEQ + kt + cg);
    asm volatile("global_load_async_to_lds_b128 %0, %1, off" :: "v"(ldsK[buf]), "v"(gK) : "memory");
    asm volatile("global_load_async_to_lds_b128 %0, %1, off" :: "v"(ldsV[buf]), "v"(gV) : "memory");
  };
  auto compute = [&](int buf) {
    const int l15 = lane & 15, hi = lane >> 4;
    FragBf a, fb0, fb1;
    a.h[0]  = *(const v8bf*)&Ks[buf][wm*16 + l15][hi*8];
    a.h[1]  = *(const v8bf*)&Ks[buf][wm*16 + l15][hi*8 + 16];
    fb0.h[0] = *(const v8bf*)&Vs[buf][(wn+0)*16 + l15][hi*8];
    fb0.h[1] = *(const v8bf*)&Vs[buf][(wn+0)*16 + l15][hi*8 + 16];
    fb1.h[0] = *(const v8bf*)&Vs[buf][(wn+1)*16 + l15][hi*8];
    fb1.h[1] = *(const v8bf*)&Vs[buf][(wn+1)*16 + l15][hi*8 + 16];
    c0 = __builtin_amdgcn_wmma_f32_16x16x32_bf16(false, a.v, false, fb0.v, (short)0, c0, false, false);
    c1 = __builtin_amdgcn_wmma_f32_16x16x32_bf16(false, a.v, false, fb1.v, (short)0, c1, false, false);
  };

  const int k_begin = split * (SEQ / 8);
  const int NSTEP = (SEQ / 8) / BK;    // 16
  issue_async(0, k_begin);
  int p = 0;
  for (int i = 0; i < NSTEP; ++i) {
    asm volatile("s_wait_asynccnt 0x0" ::: "memory");   // buf p ready (this wave)
    __syncthreads();                                    // all waves' copies ready
    if (i + 1 < NSTEP) issue_async(1 - p, k_begin + (i + 1) * BK);
    compute(p);
    p ^= 1;
  }

  float* Cb = PART + ((size_t)bh * 8 + split) * 4096;
  const int l15 = lane & 15, hi8 = (lane >> 4) * 8;
  #pragma unroll
  for (int i = 0; i < 8; ++i) {
    int d = wm*16 + hi8 + i;
    Cb[d*64 + (wn+0)*16 + l15] = c0[i];
    Cb[d*64 + (wn+1)*16 + l15] = c1[i];
  }
}

__global__ __launch_bounds__(256)
void k_ctx_reduce(const float* __restrict__ PART, float* __restrict__ CTX)
{
  int bh = blockIdx.x, tid = threadIdx.x;
  for (int i = tid; i < 4096; i += 256) {
    float s = 0.f;
    #pragma unroll
    for (int sp = 0; sp < 8; ++sp) s += PART[((size_t)bh * 8 + sp) * 4096 + i];
    CTX[(size_t)bh * 4096 + i] = s;
  }
}

// ---------------------------------------------------------------------------
// W2[b][o][h*64+d] = sum_e Wout[o][h*64+e] * ctx[b,h][d][e]  -> bf16
// ---------------------------------------------------------------------------
__global__ __launch_bounds__(256)
void k_build_w2(const float* __restrict__ Wout, const float* __restrict__ CTX,
                __bf16* __restrict__ W2)
{
  __shared__ float ctx[64][64];
  int bh = blockIdx.x; int b = bh >> 3, h = bh & 7;
  const float* Cb = CTX + (size_t)bh * 4096;
  int tid = threadIdx.x;
  for (int i = tid; i < 4096; i += 256) ctx[i >> 6][i & 63] = Cb[i];
  __syncthreads();
  for (int oo = 0; oo < 2; ++oo) {
    int o = tid + oo * 256;
    const float* wrow = Wout + (size_t)o * DIMC + h * 64;
    float w[64];
    #pragma unroll
    for (int e = 0; e < 64; ++e) w[e] = wrow[e];
    __bf16* dst = W2 + (size_t)(b * 512 + o) * 512 + h * 64;
    for (int d = 0; d < 64; ++d) {
      float acc = 0.f;
      #pragma unroll
      for (int e = 0; e < 64; ++e) acc += w[e] * ctx[d][e];
      dst[d] = (__bf16)acc;
    }
  }
}

// ---------------------------------------------------------------------------
// GEMM2: out[b] = W2[b](512x512 bf16) @ q'[b](512x4096) + bias ; fp32 out
// A tile (128x32, straight strided copy + LDS row pad) moved by the TDM.
// ---------------------------------------------------------------------------
__global__ __launch_bounds__(256)
void k_gemm_out(const __bf16* __restrict__ W2, const __bf16* __restrict__ QKV,
                const float* __restrict__ bias, float* __restrict__ Out)
{
  __shared__ __bf16 As[2][128][APITCH];
  __shared__ __bf16 Bs[2][64][APITCH];
  const int tid = threadIdx.x, lane = tid & 31, wave = tid >> 5;
  const int n0 = blockIdx.x * 64;
  const int m0 = blockIdx.y * 128;
  const int b  = blockIdx.z;
  const __bf16* Ab = W2  + (size_t)b * 512 * 512;
  const __bf16* Qb = QKV + (size_t)b * QKV_ROWS * SEQ;   // q' = rows 0..511
  float* Ob = Out + (size_t)b * DIMC * SEQ;
  const int wm2 = (wave & 3) * 2, wn2 = (wave >> 2) * 2;
  v8f c00 = {}, c01 = {}, c10 = {}, c11 = {};
  const int kp = tid >> 4, ng = (tid & 15) * 4;
  const uint32_t ldsA[2] = { lds_off(&As[0][0][0]), lds_off(&As[1][0][0]) };

  // TDM D# for a 2D 128(row) x 32(col) bf16 tile of the 512x512 W2[b] tensor,
  // LDS-padded 4 DWORDs every 16 DWORDs -> 80B row pitch == APITCH.
  auto tdm_issue = [&](int buf, int kt) {
    unsigned long long ga = (unsigned long long)(uintptr_t)(Ab + (size_t)m0 * 512 + kt);
    v4u g0;
    g0[0] = 1u;                                            // count=1, user mode
    g0[1] = ldsA[buf];                                     // lds_addr
    g0[2] = (uint32_t)ga;                                  // global_addr lo
    g0[3] = (uint32_t)((ga >> 32) & 0x01FFFFFFu) | (2u << 30);  // addr hi | type=2
    v8u g1;
    g1[0] = (1u << 16)    // data_size = 2B
          | (1u << 20)    // pad_enable
          | (3u << 22)    // pad_interval: 16 DWORDs
          | (3u << 25);   // pad_amount:   4 DWORDs
    g1[1] = (512u & 0xffffu) << 16;        // tensor_dim0[15:0]=512
    g1[2] = (512u & 0xffffu) << 16;        // dim0 hi=0 | tensor_dim1[15:0]=512
    g1[3] = (32u << 16);                   // dim1 hi=0 | tile_dim0=32
    g1[4] = 128u;                          // tile_dim1=128 | tile_dim2=0
    g1[5] = 512u;                          // tensor_dim0_stride lo
    g1[6] = 0u;
    g1[7] = 0u;
    asm volatile("tensor_load_to_lds %0, %1" :: "s"(g0), "s"(g1) : "memory");
  };

  auto load_b = [&](int buf, int kt) {  // bf16 transpose, packed b32 stores
    uint64_t r0 = *(const uint64_t*)(Qb + (size_t)(kt + 2*kp)     * SEQ + n0 + ng);
    uint64_t r1 = *(const uint64_t*)(Qb + (size_t)(kt + 2*kp + 1) * SEQ + n0 + ng);
    #pragma unroll
    for (int j = 0; j < 4; ++j) {
      uint32_t w = (uint32_t)((r0 >> (16*j)) & 0xffffu) |
                   ((uint32_t)((r1 >> (16*j)) & 0xffffu) << 16);
      *(uint32_t*)&Bs[buf][ng + j][2*kp] = w;
    }
  };

  auto compute = [&](int buf) {
    const int l15 = lane & 15, hi = lane >> 4;
    FragBf a0, a1, fb0, fb1;
    a0.h[0] = *(const v8bf*)&As[buf][(wm2+0)*16 + l15][hi*8];
    a0.h[1] = *(const v8bf*)&As[buf][(wm2+0)*16 + l15][hi*8 + 16];
    a1.h[0] = *(const v8bf*)&As[buf][(wm2+1)*16 + l15][hi*8];
    a1.h[1] = *(const v8bf*)&As[buf][(wm2+1)*16 + l15][hi*8 + 16];
    fb0.h[0] = *(const v8bf*)&Bs[buf][(wn2+0)*16 + l15][hi*8];
    fb0.h[1] = *(const v8bf*)&Bs[buf][(wn2+0)*16 + l15][hi*8 + 16];
    fb1.h[0] = *(const v8bf*)&Bs[buf][(wn2+1)*16 + l15][hi*8];
    fb1.h[1] = *(const v8bf*)&Bs[buf][(wn2+1)*16 + l15][hi*8 + 16];
    c00 = __builtin_amdgcn_wmma_f32_16x16x32_bf16(false, a0.v, false, fb0.v, (short)0, c00, false, false);
    c01 = __builtin_amdgcn_wmma_f32_16x16x32_bf16(false, a0.v, false, fb1.v, (short)0, c01, false, false);
    c10 = __builtin_amdgcn_wmma_f32_16x16x32_bf16(false, a1.v, false, fb0.v, (short)0, c10, false, false);
    c11 = __builtin_amdgcn_wmma_f32_16x16x32_bf16(false, a1.v, false, fb1.v, (short)0, c11, false, false);
  };

  if (tid < 32) tdm_issue(0, 0);       // wave 0 drives the TDM
  load_b(0, 0);
  int p = 0;
  const int NSTEP = DIMC / BK;         // 16
  for (int i = 0; i < NSTEP; ++i) {
    if (tid < 32) __builtin_amdgcn_s_wait_tensorcnt((short)0);  // buf p A-tile landed
    __syncthreads();
    if (i + 1 < NSTEP) {
      if (tid < 32) tdm_issue(1 - p, (i + 1) * BK);
      load_b(1 - p, (i + 1) * BK);
    }
    compute(p);
    p ^= 1;
  }

  const int l15 = lane & 15, hi8 = (lane >> 4) * 8;
  #pragma unroll
  for (int i = 0; i < 8; ++i) {
    int r0 = m0 + (wm2+0)*16 + hi8 + i;
    int r1 = m0 + (wm2+1)*16 + hi8 + i;
    float bv0 = bias[r0], bv1 = bias[r1];
    Ob[(size_t)r0 * SEQ + n0 + (wn2+0)*16 + l15] = c00[i] + bv0;
    Ob[(size_t)r0 * SEQ + n0 + (wn2+1)*16 + l15] = c01[i] + bv0;
    Ob[(size_t)r1 * SEQ + n0 + (wn2+0)*16 + l15] = c10[i] + bv1;
    Ob[(size_t)r1 * SEQ + n0 + (wn2+1)*16 + l15] = c11[i] + bv1;
  }
}

// ---------------------------------------------------------------------------
// RMSNorm over channel dim (512) per (b,n) column, in-place on Out (L2-resident)
// ---------------------------------------------------------------------------
__global__ __launch_bounds__(256)
void k_rmsnorm(float* __restrict__ Out, const float* __restrict__ g)
{
  int t = blockIdx.x * blockDim.x + threadIdx.x;   // b*SEQ + n
  int n = t & (SEQ - 1), b = t >> 12;
  float* base = Out + (size_t)b * DIMC * SEQ + n;
  float ss = 0.f;
  for (int c = 0; c < DIMC; ++c) { float v = base[(size_t)c * SEQ]; ss += v * v; }
  float norm = sqrtf(ss);
  float scale = 22.62741699796952f / fmaxf(norm, 1e-12f);   // sqrt(512)/norm
  for (int c = 0; c < DIMC; ++c) base[(size_t)c * SEQ] = base[(size_t)c * SEQ] * scale * g[c];
}

// ---------------------------------------------------------------------------
extern "C" void kernel_launch(void* const* d_in, const int* in_sizes, int n_in,
                              void* d_out, int out_size, void* d_ws, size_t ws_size,
                              hipStream_t stream)
{
  const float* x     = (const float*)d_in[0];
  const float* w_qkv = (const float*)d_in[1];
  const float* w_out = (const float*)d_in[2];
  const float* b_out = (const float*)d_in[3];
  const float* g     = (const float*)d_in[4];
  float* out = (float*)d_out;

  char* ws = (char*)d_ws;
  __bf16* qkv  = (__bf16*)ws;                          // 8*1536*4096*2 = 100,663,296 B
  float*  part = (float*)(ws + 100663296);             // 64*8*4096*4   =   8,388,608 B
  float*  ctx  = (float*)(ws + 100663296 + 8388608);   // 64*4096*4     =   1,048,576 B
  __bf16* w2   = (__bf16*)(ws + 100663296 + 8388608 + 1048576); // 8*512*512*2 = 4,194,304 B

  k_gemm_qkv<<<dim3(SEQ / 64, QKV_ROWS / 128, BATCH), 256, 0, stream>>>(w_qkv, x, qkv);
  k_softmax_q<<<dim3(BATCH * HEADS * SEQ / 256), 256, 0, stream>>>(qkv);
  k_softmax_k<<<dim3(BATCH * DIMC), 256, 0, stream>>>(qkv);
  k_context<<<dim3(BATCH * HEADS * 8), 256, 0, stream>>>(qkv, part);
  k_ctx_reduce<<<dim3(BATCH * HEADS), 256, 0, stream>>>(part, ctx);
  k_build_w2<<<dim3(BATCH * HEADS), 256, 0, stream>>>(w_out, ctx, w2);
  k_gemm_out<<<dim3(SEQ / 64, DIMC / 128, BATCH), 256, 0, stream>>>(w2, qkv, b_out, out);
  k_rmsnorm<<<dim3(BATCH * SEQ / 256), 256, 0, stream>>>(out, g);
}
